// TreeDecoder_16458314678306
// MI455X (gfx1250) — compile-verified, matrix operand order checked by reading, not run on hardware
//
#include <hip/hip_runtime.h>
#include <stdint.h>

// ---------------------------------------------------------------------------
// Types
// ---------------------------------------------------------------------------
typedef __attribute__((ext_vector_type(16))) __bf16          v16bf;
typedef __attribute__((ext_vector_type(16))) unsigned short  v16us;
typedef __attribute__((ext_vector_type(8)))  float           v8f;
typedef __attribute__((ext_vector_type(4)))  unsigned int    v4u;
typedef __attribute__((ext_vector_type(8)))  int             v8i;
typedef __attribute__((ext_vector_type(4)))  int             v4i;

#if defined(__has_builtin)
#  if __has_builtin(__builtin_amdgcn_tensor_load_to_lds) && __has_builtin(__builtin_amdgcn_s_wait_tensorcnt)
#    define HAVE_TDM 1
#  else
#    define HAVE_TDM 0
#  endif
#else
#  define HAVE_TDM 0
#endif

// ---------------------------------------------------------------------------
// bf16 helpers (stored as ushort bits)
// ---------------------------------------------------------------------------
static __device__ __forceinline__ unsigned short f2bf(float f) {
    unsigned u = __float_as_uint(f);
    u += 0x7FFFu + ((u >> 16) & 1u);      // round-to-nearest-even-ish
    return (unsigned short)(u >> 16);
}
static __device__ __forceinline__ float bf2f(unsigned short h) {
    return __uint_as_float(((unsigned)h) << 16);
}

// ---------------------------------------------------------------------------
// TDM: async 2D tile load (tile = 32 x 64 bf16) from strided tensor into LDS.
// D# per CDNA5 ISA:
//   group0: [1:0]=count=1, [63:32]=lds_addr, [120:64]=global_addr, [127:126]=2
//   group1: data_size=1(2B), tensor_dim0, tensor_dim1, tile 32x64, dim0 stride
// ---------------------------------------------------------------------------
#if HAVE_TDM
static __device__ __forceinline__ void tdm_load_tile_2d(
        unsigned lds_off, unsigned long long gaddr,
        unsigned tensor_w, unsigned tensor_h, unsigned stride_elems) {
    v4u g0; v8i g1;
    v4i z4 = {0, 0, 0, 0};
    g0[0] = 1u;                                                  // count=1, user desc
    g0[1] = lds_off;                                             // lds_addr
    g0[2] = (unsigned)(gaddr & 0xFFFFFFFFull);                   // global_addr lo
    g0[3] = ((unsigned)(gaddr >> 32) & 0x01FFFFFFu) | 0x80000000u; // addr hi | type=2
    g1[0] = 0x00010000;                                          // data_size = 2 bytes
    g1[1] = (int)((tensor_w & 0xFFFFu) << 16);                   // tensor_dim0 lo
    g1[2] = (int)((tensor_w >> 16) | ((tensor_h & 0xFFFFu) << 16));
    g1[3] = (int)((tensor_h >> 16) | (32u << 16));               // tile_dim0 = 32
    g1[4] = 64;                                                  // tile_dim1=64, tile_dim2=0
    g1[5] = (int)stride_elems;                                   // dim0_stride lo (<=1152)
    g1[6] = 0;
    g1[7] = 0;
#if __clang_major__ >= 23
    v8i z8 = {0, 0, 0, 0, 0, 0, 0, 0};
    __builtin_amdgcn_tensor_load_to_lds(g0, g1, z4, z4, z8, 0);
#else
    __builtin_amdgcn_tensor_load_to_lds(g0, g1, z4, z4, 0);
#endif
}
#endif

// ---------------------------------------------------------------------------
// Weight conversion: f32 [O,Kreal] -> bf16 [O,Kpad] (zero padded K)
// ---------------------------------------------------------------------------
__global__ void convert_w(const float* __restrict__ w, unsigned short* __restrict__ dst,
                          int Kreal, int Kpad, int total) {
    int i = blockIdx.x * blockDim.x + threadIdx.x;
    if (i >= total) return;
    int o = i / Kpad, k = i - o * Kpad;
    dst[i] = (k < Kreal) ? f2bf(w[o * Kreal + k]) : (unsigned short)0;
}

__global__ void root_init(const float* __restrict__ root, unsigned short* __restrict__ H) {
    int i = blockIdx.x * blockDim.x + threadIdx.x;
    if (i < 8 * 128) H[i] = f2bf(root[i & 127]);
}

__global__ void fill_zero_u16(unsigned short* __restrict__ p, int n) {
    int i = blockIdx.x * blockDim.x + threadIdx.x;
    if (i < n) p[i] = 0;
}

// ---------------------------------------------------------------------------
// Prep: build X[M,320] bf16 = [ h(128) | layernorm(emb)(128) | pos(39) pad->64 ]
// One wave32 per node.
// ---------------------------------------------------------------------------
static __device__ __forceinline__ unsigned dein(unsigned x) {
    x &= 0x55555555u;
    x = (x | (x >> 1)) & 0x33333333u;
    x = (x | (x >> 2)) & 0x0F0F0F0Fu;
    x = (x | (x >> 4)) & 0x00FF00FFu;
    x = (x | (x >> 8)) & 0x0000FFFFu;
    return x;
}

__global__ __launch_bounds__(256)
void prep_kernel(const unsigned short* __restrict__ H, const float* __restrict__ emb,
                 const float* __restrict__ gw, const float* __restrict__ gb,
                 const int* __restrict__ keys, int depth, int M,
                 unsigned short* __restrict__ X) {
    const int lane = threadIdx.x & 31;
    const int node = blockIdx.x * 8 + (threadIdx.x >> 5);
    if (node >= M) return;

    // --- copy h (bf16 -> bf16)
    #pragma unroll
    for (int i = 0; i < 4; ++i) {
        int c = lane * 4 + i;
        X[node * 320 + c] = H[node * 128 + c];
    }

    // --- layernorm(emb)
    float e[4]; float s = 0.f, s2 = 0.f;
    #pragma unroll
    for (int i = 0; i < 4; ++i) {
        e[i] = emb[node * 128 + lane * 4 + i];
        s += e[i]; s2 += e[i] * e[i];
    }
    #pragma unroll
    for (int off = 16; off >= 1; off >>= 1) {
        s  += __shfl_xor(s,  off, 32);
        s2 += __shfl_xor(s2, off, 32);
    }
    float mu  = s * (1.0f / 128.0f);
    float var = s2 * (1.0f / 128.0f) - mu * mu;
    float inv = rsqrtf(var + 1e-5f);
    #pragma unroll
    for (int i = 0; i < 4; ++i) {
        int c = lane * 4 + i;
        X[node * 320 + 128 + c] = f2bf((e[i] - mu) * inv * gw[c] + gb[c]);
    }

    // --- positional Fourier features (39 real, pad to 64)
    unsigned key = (unsigned)keys[node];
    float res = (float)(1u << depth);
    float bx = ((float)dein(key)      + 0.5f) / res;
    float by = ((float)dein(key >> 1) + 0.5f) / res;
    float bd = (float)depth * (1.0f / 7.0f);
    float base[3] = {bx, by, bd};
    #pragma unroll
    for (int t = 0; t < 2; ++t) {
        int p = lane * 2 + t;
        float v = 0.f;
        if (p < 3) {
            v = base[p];
        } else if (p < 39) {
            int a = (p - 3) / 12, w = (p - 3) % 12;
            int f = (w < 6) ? w : (w - 6);
            float ang = base[a] * 6.28318530717958647692f * (float)(1 << f);
            v = (w < 6) ? __sinf(ang) : __cosf(ang);
        }
        X[node * 320 + 256 + p] = f2bf(v);
    }
}

// ---------------------------------------------------------------------------
// WMMA GEMM:  Out[M,Nout] = op(A[M,K]) @ W[Nout,K]^T + bias  (bf16 in, bf16 out)
//   GATHER : A[m,k] gathered from Hsrc via neighs[m*9 + k/128] (k%128 column)
//   SCATTER: output column group c=n/128 scattered to Hnext[children[m*4+c]]
// Block: 256 threads = 8 wave32s; 64x64 tile; wave = 16x32 (2 WMMAs / K-step).
// Tiles staged by the Tensor Data Mover (wave 0 issues, TENSORcnt-synced),
// double-buffered in LDS so the DMA overlaps the matrix pipe.
// ---------------------------------------------------------------------------
template<int GATHER, int SCATTER, int RELU>
__global__ __launch_bounds__(256)
void gemm_kernel(const unsigned short* __restrict__ A, int lda,
                 const int* __restrict__ neighs,
                 const unsigned short* __restrict__ W, int K,
                 const float* __restrict__ bias,
                 int M, int Nout,
                 unsigned short* __restrict__ Out, int ldo,
                 const int* __restrict__ children, int Nnext) {
    __shared__ __align__(16) unsigned short As[2][64 * 32];
    __shared__ __align__(16) unsigned short Bs[2][64 * 32];

    const int tid  = threadIdx.x;
    const int lane = tid & 31;
    const int wave = tid >> 5;
    const int wm   = wave & 3;    // 0..3 : which 16-row slab
    const int wn   = wave >> 2;   // 0..1 : which 32-col slab
    const int mBase = blockIdx.x * 64;
    const int nBase = blockIdx.y * 64;

    const int loadRow = tid >> 2;        // 0..63
    const int loadK   = (tid & 3) * 8;   // 0,8,16,24
    const int half = lane >> 4, l16 = lane & 15;

    v8f acc0 = {}; v8f acc1 = {};

    auto stageB = [&](int buf, int k0) {
#if HAVE_TDM
        if (wave == 0)
            tdm_load_tile_2d((unsigned)(uintptr_t)&Bs[buf][0],
                             (unsigned long long)(uintptr_t)W +
                                 ((unsigned long long)nBase * (unsigned)K + (unsigned)k0) * 2ull,
                             32u, (unsigned)(Nout - nBase), (unsigned)K);
#else
        int n = nBase + loadRow;
        uint4 bv = *(const uint4*)(W + (size_t)n * K + k0 + loadK);
        *(uint4*)(&Bs[buf][loadRow * 32 + loadK]) = bv;
#endif
    };
    auto stageA = [&](int buf, int k0) {
        if (GATHER) {
            int m = mBase + loadRow;
            #pragma unroll
            for (int i = 0; i < 8; ++i) {
                int k = k0 + loadK + i;
                unsigned short v = 0;
                if (m < M) {
                    int j = k >> 7, c = k & 127;
                    int idx = neighs[m * 9 + j];
                    if (idx >= 0 && idx < M) v = A[idx * 128 + c];
                }
                As[buf][loadRow * 32 + loadK + i] = v;
            }
        } else {
#if HAVE_TDM
            if (wave == 0)
                tdm_load_tile_2d((unsigned)(uintptr_t)&As[buf][0],
                                 (unsigned long long)(uintptr_t)A +
                                     ((unsigned long long)mBase * (unsigned)lda + (unsigned)k0) * 2ull,
                                 32u, (unsigned)(M - mBase), (unsigned)lda);
#else
            int m = mBase + loadRow;
            uint4 v = make_uint4(0u, 0u, 0u, 0u);
            if (m < M) v = *(const uint4*)(A + (size_t)m * lda + k0 + loadK);
            *(uint4*)(&As[buf][loadRow * 32 + loadK]) = v;
#endif
        }
    };

    // ---- prologue: fill buffer 0
    stageA(0, 0);
    stageB(0, 0);
#if HAVE_TDM
    if (wave == 0) __builtin_amdgcn_s_wait_tensorcnt(0);
#endif
    __syncthreads();

    int cur = 0;
    for (int k0 = 0; k0 < K; k0 += 32) {
        const int nxt = cur ^ 1;
        if (k0 + 32 < K) {               // kick off next tile's DMA
            stageA(nxt, k0 + 32);
            stageB(nxt, k0 + 32);
        }

        // ---- fragments per ISA 7.12.2 (16-bit A 16x32 layout); B mirrored
        v16us au, bu0, bu1;
        #pragma unroll
        for (int j = 0; j < 16; ++j) {
            int kk = (j < 8) ? (half * 8 + j) : (8 + half * 8 + j); // = 16+half*8+(j-8)
            au[j]  = As[cur][(wm * 16      + l16) * 32 + kk];
            bu0[j] = Bs[cur][(wn * 32      + l16) * 32 + kk];
            bu1[j] = Bs[cur][(wn * 32 + 16 + l16) * 32 + kk];
        }
        v16bf a  = __builtin_bit_cast(v16bf, au);
        v16bf b0 = __builtin_bit_cast(v16bf, bu0);
        v16bf b1 = __builtin_bit_cast(v16bf, bu1);
        acc0 = __builtin_amdgcn_wmma_f32_16x16x32_bf16(false, a, false, b0, (short)0, acc0, false, false);
        acc1 = __builtin_amdgcn_wmma_f32_16x16x32_bf16(false, a, false, b1, (short)0, acc1, false, false);

        __syncthreads();                  // everyone done reading 'cur'
#if HAVE_TDM
        if (wave == 0) __builtin_amdgcn_s_wait_tensorcnt(0);  // 'nxt' landed
#endif
        __syncthreads();                  // publish 'nxt'
        cur = nxt;
    }

    // ---- epilogue (C/D layout: VGPR r -> row r + 8*half, col = lane&15)
    #pragma unroll
    for (int r = 0; r < 8; ++r) {
        int row = mBase + wm * 16 + half * 8 + r;
        if (row >= M) continue;
        #pragma unroll
        for (int s = 0; s < 2; ++s) {
            int colG = nBase + wn * 32 + s * 16 + l16;
            float v = (s ? acc1[r] : acc0[r]) + bias[colG];
            if (RELU) v = fmaxf(v, 0.0f);
            if (SCATTER) {
                int c = colG >> 7, cc = colG & 127;
                int dst = children[row * 4 + c];
                if (dst >= 0 && dst < Nnext) Out[(size_t)dst * 128 + cc] = f2bf(v);
            } else {
                Out[(size_t)row * ldo + colG] = f2bf(v);
            }
        }
    }
}

// ---------------------------------------------------------------------------
// 128 -> 1 head: out[node] = dot(bf16 H row, f32 w) + b   (one wave per node)
// ---------------------------------------------------------------------------
__global__ __launch_bounds__(256)
void head_dot(const unsigned short* __restrict__ H, const float* __restrict__ w,
              const float* __restrict__ b, float* __restrict__ out, int M) {
    const int lane = threadIdx.x & 31;
    const int node = blockIdx.x * 8 + (threadIdx.x >> 5);
    if (node >= M) return;
    float s = 0.f;
    #pragma unroll
    for (int i = 0; i < 4; ++i) {
        int c = lane * 4 + i;
        s += bf2f(H[node * 128 + c]) * w[c];
    }
    #pragma unroll
    for (int off = 16; off >= 1; off >>= 1) s += __shfl_xor(s, off, 32);
    if (lane == 0) out[node] = s + b[0];
}

// ---------------------------------------------------------------------------
// Host orchestration
// ---------------------------------------------------------------------------
extern "C" void kernel_launch(void* const* d_in, const int* in_sizes, int n_in,
                              void* d_out, int out_size, void* d_ws, size_t ws_size,
                              hipStream_t stream) {
    (void)in_sizes; (void)n_in; (void)out_size; (void)ws_size;
    static const int Ns[8] = {8, 32, 128, 512, 2048, 8192, 32768, 131072};
    const int MAXN = 131072;

    // ---- input pointer map (setup_inputs flattening order)
    const float* embs[8];
    for (int d = 0; d < 8; ++d) embs[d] = (const float*)d_in[d];
    const float* rootTok = (const float*)d_in[8];
    const float* fuse1w = (const float*)d_in[9];  const float* fuse1b = (const float*)d_in[10];
    const float* fuse2w = (const float*)d_in[11]; const float* fuse2b = (const float*)d_in[12];
    const float* skipW  = (const float*)d_in[13]; const float* skipB  = (const float*)d_in[14];
    const float* sp1w = (const float*)d_in[15]; const float* sp1b = (const float*)d_in[16];
    const float* sp2w = (const float*)d_in[17]; const float* sp2b = (const float*)d_in[18];
    const float* ch1w = (const float*)d_in[19]; const float* ch1b = (const float*)d_in[20];
    const float* ch2w = (const float*)d_in[21]; const float* ch2b = (const float*)d_in[22];
    const float* v1w  = (const float*)d_in[23]; const float* v1b  = (const float*)d_in[24];
    const float* v2w  = (const float*)d_in[25]; const float* v2b  = (const float*)d_in[26];
    const float* mixW[8]; const float* mixB[8];
    for (int d = 0; d < 8; ++d) { mixW[d] = (const float*)d_in[27 + 2*d]; mixB[d] = (const float*)d_in[28 + 2*d]; }
    const int* keys[8];  for (int d = 0; d < 8; ++d) keys[d]  = (const int*)d_in[43 + d];
    const int* neigh[8]; for (int d = 0; d < 8; ++d) neigh[d] = (const int*)d_in[51 + d];
    const int* child[7]; for (int d = 0; d < 7; ++d) child[d] = (const int*)d_in[59 + d];
    float* out = (float*)d_out;

    // ---- output offsets: split_logits d=0..6 then val_pred d=0..7
    int splitOff[7], valOff[8], a = 0;
    for (int d = 0; d < 7; ++d) { splitOff[d] = a; a += Ns[d]; }
    for (int d = 0; d < 8; ++d) { valOff[d]  = a; a += Ns[d]; }

    // ---- workspace carve-out (256B aligned)
    char* base = (char*)d_ws; size_t off = 0;
    auto alloc = [&](size_t bytes) -> void* {
        void* p = base + off;
        off = (off + bytes + 255) & ~(size_t)255;
        return p;
    };
    unsigned short* Wf1  = (unsigned short*)alloc((size_t)128 * 320 * 2);
    unsigned short* Wf2  = (unsigned short*)alloc((size_t)128 * 128 * 2);
    unsigned short* Wsp1 = (unsigned short*)alloc((size_t)128 * 128 * 2);
    unsigned short* Wch1 = (unsigned short*)alloc((size_t)128 * 128 * 2);
    unsigned short* Wch2 = (unsigned short*)alloc((size_t)512 * 128 * 2);
    unsigned short* Wv1  = (unsigned short*)alloc((size_t)128 * 128 * 2);
    unsigned short* Wmix[8];
    for (int d = 0; d < 8; ++d) Wmix[d] = (unsigned short*)alloc((size_t)128 * 1152 * 2);
    unsigned short* X  = (unsigned short*)alloc((size_t)MAXN * 320 * 2);
    unsigned short* T1 = (unsigned short*)alloc((size_t)MAXN * 128 * 2);
    unsigned short* T2 = (unsigned short*)alloc((size_t)MAXN * 128 * 2);
    unsigned short* Ha = (unsigned short*)alloc((size_t)MAXN * 128 * 2);
    unsigned short* Hb = (unsigned short*)alloc((size_t)MAXN * 128 * 2);

    // ---- convert weights to padded bf16
    auto conv = [&](const float* w, unsigned short* dst, int O, int Kreal, int Kpad) {
        int total = O * Kpad;
        convert_w<<<(total + 255) / 256, 256, 0, stream>>>(w, dst, Kreal, Kpad, total);
    };
    conv(fuse1w, Wf1, 128, 295, 320);
    conv(fuse2w, Wf2, 128, 128, 128);
    conv(sp1w,  Wsp1, 128, 128, 128);
    conv(ch1w,  Wch1, 128, 128, 128);
    conv(ch2w,  Wch2, 512, 128, 128);
    conv(v1w,   Wv1,  128, 128, 128);
    for (int d = 0; d < 8; ++d) conv(mixW[d], Wmix[d], 128, 1152, 1152);

    // ---- h0 = broadcast(root_token) in bf16
    root_init<<<4, 256, 0, stream>>>(rootTok, Ha);

    auto gemmDirect = [&](const unsigned short* A_, const unsigned short* W_, int K,
                          const float* bias_, int M, int Nout, bool relu,
                          unsigned short* Out_) {
        dim3 g((M + 63) / 64, Nout / 64);
        if (relu)
            gemm_kernel<0, 0, 1><<<g, 256, 0, stream>>>(A_, K, nullptr, W_, K, bias_,
                                                        M, Nout, Out_, Nout, nullptr, 0);
        else
            gemm_kernel<0, 0, 0><<<g, 256, 0, stream>>>(A_, K, nullptr, W_, K, bias_,
                                                        M, Nout, Out_, Nout, nullptr, 0);
    };

    for (int d = 0; d < 8; ++d) {
        const int M = Ns[d];
        const int nodeBlocks = (M + 7) / 8;

        // X = [h | layernorm(emb) | fourier(pos)]  (K padded to 320)
        prep_kernel<<<nodeBlocks, 256, 0, stream>>>(Ha, embs[d], skipW, skipB,
                                                    keys[d], d, M, X);
        // fuse: relu(X @ Wf1^T + b1) -> T1 ; T1 @ Wf2^T + b2 -> (Ha | T2)
        gemmDirect(X, Wf1, 320, fuse1b, M, 128, true, T1);
        unsigned short* fuseOut = (d == 0) ? Ha : T2;
        gemmDirect(T1, Wf2, 128, fuse2b, M, 128, false, fuseOut);

        if (d >= 1) {
            // QuadConv: relu(gather9(T2) @ Wmix^T + b) -> Ha  (gather fused in A load)
            dim3 g((M + 63) / 64, 128 / 64);
            gemm_kernel<1, 0, 1><<<g, 256, 0, stream>>>(T2, 128, neigh[d], Wmix[d], 1152,
                                                        mixB[d], M, 128, Ha, 128,
                                                        nullptr, 0);
        }

        // val head
        gemmDirect(Ha, Wv1, 128, v1b, M, 128, true, T1);
        head_dot<<<nodeBlocks, 256, 0, stream>>>(T1, v2w, v2b, out + valOff[d], M);

        if (d < 7) {
            // split head
            gemmDirect(Ha, Wsp1, 128, sp1b, M, 128, true, T1);
            head_dot<<<nodeBlocks, 256, 0, stream>>>(T1, sp2w, sp2b, out + splitOff[d], M);

            // child head -> scatter into next level's h
            const int Nnext = Ns[d + 1];
            int zn = Nnext * 128;
            fill_zero_u16<<<(zn + 255) / 256, 256, 0, stream>>>(Hb, zn);
            gemmDirect(Ha, Wch1, 128, ch1b, M, 128, true, T1);
            dim3 g((M + 63) / 64, 512 / 64);
            gemm_kernel<0, 1, 0><<<g, 256, 0, stream>>>(T1, 128, nullptr, Wch2, 128,
                                                        ch2b, M, 512, Hb, 0,
                                                        child[d], Nnext);
            unsigned short* t = Ha; Ha = Hb; Hb = t;
        }
    }
}
